// TransitionUp_1984274891514
// MI455X (gfx1250) — matrix operand order repaired and checked by reading, not compiled, not run on hardware
//
#include <hip/hip_runtime.h>
#include <hip/hip_bf16.h>

// ---------------------------------------------------------------------------
// TransitionUp (PointNet++-style): mlp(x_sub) -> knn_interpolate(k=3) -> +mlp(x)
// MI455X / gfx1250: wave32, WMMA f32_16x16x32_f16 for both GEMMs.
//  - W pre-converted to f16 transposed once, then LDS-resident per block
//  - 64x128 block tile, 4 v_wmma per wave per k-step (fully unrolled K)
//  - coarse point cloud LDS-resident (48KB of 320KB WGP LDS) for the KNN scan
// ---------------------------------------------------------------------------

typedef __attribute__((ext_vector_type(16))) _Float16 v16h;
typedef __attribute__((ext_vector_type(8)))  _Float16 v8h;
typedef __attribute__((ext_vector_type(8)))  float    v8f;

#define NCOL 128   // both MLP output widths are 128
#define EPS_BN 1e-5f
#define EPS_W  1e-16f

#define A_STRIDE 40          // 64x32 A tile padded to 40 halves/row (bank spread)
#define WT_PAD   8           // Wt rows padded by 8 halves (bank spread)

// ---------------------------------------------------------------------------
// One-time W conversion: wt[n*K + k] = (f16) W[k*128 + n]   (transposed f16)
// ---------------------------------------------------------------------------
__global__ void convert_w_kernel(const float* __restrict__ Wm,
                                 _Float16* __restrict__ wt, int Kdim)
{
    int l = blockIdx.x * blockDim.x + threadIdx.x;   // over 128*Kdim
    int n = l / Kdim;
    int k = l - n * Kdim;
    wt[l] = (_Float16)Wm[(size_t)k * NCOL + n];
}

// ---------------------------------------------------------------------------
// Fused GEMM (H = X @ W + bias) + batchnorm statistics (column sum / sumsq).
// X: [M, KDIM] f32 row-major; wt: [128, KDIM] f16 (transposed W).
// Block = 256 threads = 8 waves computing a 64(M) x 128(N) slab:
//   wave wv owns columns [16*wv, 16*wv+16) and all 4 row tiles (4 accums).
// Whole Wt lives in LDS for the entire kernel; k-loop restages only A.
// ---------------------------------------------------------------------------
template <int KDIM>
__global__ __launch_bounds__(256) void gemm_bn_kernel(
    const float* __restrict__ X, const _Float16* __restrict__ wt,
    const float* __restrict__ bias, float* __restrict__ H,
    float* __restrict__ gsum, float* __restrict__ gsq, int M)
{
    __shared__ alignas(16) _Float16 lds_Wt[128 * (KDIM + WT_PAD)];
    __shared__ alignas(16) _Float16 lds_A[64 * A_STRIDE];
    __shared__ float s_sum[128];
    __shared__ float s_sq[128];

    const int t    = threadIdx.x;
    const int lane = t & 31;
    const int wv   = t >> 5;
    const int half = lane >> 4;   // 0: lanes 0-15, 1: lanes 16-31
    const int nloc = lane & 15;
    const int m0   = blockIdx.x * 64;

    if (t < 128) { s_sum[t] = 0.0f; s_sq[t] = 0.0f; }

    // Stage the full transposed-f16 W into LDS once (v8h copies, padded rows).
    for (int l = t; l < 128 * (KDIM / 8); l += 256) {
        int n  = l / (KDIM / 8);
        int kc = (l - n * (KDIM / 8)) * 8;
        v8h w8 = *(const v8h*)&wt[(size_t)n * KDIM + kc];
        *(v8h*)&lds_Wt[n * (KDIM + WT_PAD) + kc] = w8;
    }

    v8f c0 = {}, c1 = {}, c2 = {}, c3 = {};

    // A staging coords for this thread: 8 halves per thread per k-step.
    const int am = t >> 2;           // row 0..63
    const int ak = (t & 3) * 8;      // k offset 0,8,16,24
    const float* xrow = X + (size_t)(m0 + am) * KDIM + ak;

#pragma unroll
    for (int k0 = 0; k0 < KDIM; k0 += 32) {
        // Stage A tile 64x32 (f32 -> f16), one b128 store per thread.
        {
            const float4 x0 = *(const float4*)(xrow + k0);
            const float4 x1 = *(const float4*)(xrow + k0 + 4);
            v8h pk;
            pk[0] = (_Float16)x0.x; pk[1] = (_Float16)x0.y;
            pk[2] = (_Float16)x0.z; pk[3] = (_Float16)x0.w;
            pk[4] = (_Float16)x1.x; pk[5] = (_Float16)x1.y;
            pk[6] = (_Float16)x1.z; pk[7] = (_Float16)x1.w;
            *(v8h*)&lds_A[am * A_STRIDE + ak] = pk;
            if (k0 + 32 < KDIM)   // pull next k-slab toward the caches
                __builtin_prefetch(xrow + k0 + 32, 0, 3);
        }
        __syncthreads();

        // B fragment (32x16 f16): elements 0..7 -> K = k0 + 8*half + e,
        // 8..15 -> K = k0 + 16 + 8*half + (e-8); from the padded Wt row of column n.
        const _Float16* bp =
            lds_Wt + (wv * 16 + nloc) * (KDIM + WT_PAD) + k0 + half * 8;
        v8h blo = *(const v8h*)(bp);
        v8h bhi = *(const v8h*)(bp + 16);
        v16h b;
#pragma unroll
        for (int e = 0; e < 8; ++e) { b[e] = blo[e]; b[e + 8] = bhi[e]; }

        // 4 row tiles: A fragment (16x32 f16, ISA 7.12.2):
        // m = lane%16, elements 0..7 -> K = 8*half+0..7, 8..15 -> K = 16+8*half+0..7
#pragma unroll
        for (int rt = 0; rt < 4; ++rt) {
            const _Float16* ap = lds_A + (rt * 16 + nloc) * A_STRIDE + half * 8;
            v8h alo = *(const v8h*)(ap);
            v8h ahi = *(const v8h*)(ap + 16);
            v16h a;
#pragma unroll
            for (int e = 0; e < 8; ++e) { a[e] = alo[e]; a[e + 8] = ahi[e]; }

            v8f& acc = (rt == 0) ? c0 : (rt == 1) ? c1 : (rt == 2) ? c2 : c3;
            acc = __builtin_amdgcn_wmma_f32_16x16x32_f16(
                /*neg_a=*/false, a, /*neg_b=*/false, b,
                /*c_mod=*/(short)0, acc, /*reuse_a=*/false, /*reuse_b=*/false);
        }
        __syncthreads();
    }

    // Epilogue: bias add, store H, BN partial stats.
    // C/D layout: N = lane%16, VGPR r -> M = r + 8*half.
    const int col = wv * 16 + nloc;
    const float bcol = bias[col];
    float lsum = 0.0f, lsq = 0.0f;
#pragma unroll
    for (int rt = 0; rt < 4; ++rt) {
        const v8f& acc = (rt == 0) ? c0 : (rt == 1) ? c1 : (rt == 2) ? c2 : c3;
#pragma unroll
        for (int r = 0; r < 8; ++r) {
            int row = m0 + rt * 16 + half * 8 + r;
            float v = acc[r] + bcol;
            H[(size_t)row * NCOL + col] = v;
            lsum += v;
            lsq  = fmaf(v, v, lsq);
        }
    }
    atomicAdd(&s_sum[col], lsum);
    atomicAdd(&s_sq[col],  lsq);
    __syncthreads();
    if (t < 128) {
        atomicAdd(&gsum[t], s_sum[t]);
        atomicAdd(&gsq[t],  s_sq[t]);
    }
}

// ---------------------------------------------------------------------------
// Fold BN stats into per-column affine: scale = g*rsqrt(var+eps),
// shift = be - mu*scale   (so bn(h) = h*scale + shift)
// ---------------------------------------------------------------------------
__global__ void bn_finalize_kernel(const float* __restrict__ gsum,
                                   const float* __restrict__ gsq,
                                   const float* __restrict__ g,
                                   const float* __restrict__ be,
                                   float* __restrict__ scale,
                                   float* __restrict__ shift, float invM)
{
    int tc = threadIdx.x;  // 128 threads
    float mu  = gsum[tc] * invM;
    float var = gsq[tc] * invM - mu * mu;
    float inv = rsqrtf(var + EPS_BN);
    float sc  = g[tc] * inv;
    scale[tc] = sc;
    shift[tc] = fmaf(-mu, sc, be[tc]);
}

// ---------------------------------------------------------------------------
// In-place BN + ReLU (for f_sub branch).
// ---------------------------------------------------------------------------
__global__ void bn_relu_kernel(float* __restrict__ H,
                               const float* __restrict__ scale,
                               const float* __restrict__ shift, int total)
{
    int i = blockIdx.x * blockDim.x + threadIdx.x;
    if (i < total) {
        int cidx = i & (NCOL - 1);
        float v = fmaf(H[i], scale[cidx], shift[cidx]);
        H[i] = v > 0.0f ? v : 0.0f;
    }
}

// ---------------------------------------------------------------------------
// KNN (k=3) over 4096 coarse points; pos_sub cached entirely in LDS (48KB
// of the 320KB WGP LDS). One thread per fine point, register top-3, then
// normalized inverse-squared-distance weights. Uniform loop index -> LDS
// broadcast reads, conflict-free.
// ---------------------------------------------------------------------------
__global__ __launch_bounds__(256) void knn_kernel(
    const float* __restrict__ pos, const float* __restrict__ pos_sub,
    float* __restrict__ wout, int* __restrict__ iout)
{
    __shared__ float psx[4096];
    __shared__ float psy[4096];
    __shared__ float psz[4096];
    for (int j = threadIdx.x; j < 4096; j += 256) {
        psx[j] = pos_sub[j * 3 + 0];
        psy[j] = pos_sub[j * 3 + 1];
        psz[j] = pos_sub[j * 3 + 2];
    }
    __syncthreads();

    const int i = blockIdx.x * 256 + threadIdx.x;  // 64 blocks x 256 = 16384
    const float px = pos[i * 3 + 0];
    const float py = pos[i * 3 + 1];
    const float pz = pos[i * 3 + 2];

    float b0 = 3.4e38f, b1 = 3.4e38f, b2 = 3.4e38f;
    int   i0 = 0, i1 = 0, i2 = 0;
    for (int j = 0; j < 4096; ++j) {
        float dx = px - psx[j];
        float dy = py - psy[j];
        float dz = pz - psz[j];
        float d = fmaf(dx, dx, fmaf(dy, dy, dz * dz));
        if (d < b2) {
            if (d < b1) {
                if (d < b0) { b2 = b1; i2 = i1; b1 = b0; i1 = i0; b0 = d; i0 = j; }
                else        { b2 = b1; i2 = i1; b1 = d;  i1 = j; }
            } else          { b2 = d;  i2 = j; }
        }
    }
    float w0 = 1.0f / fmaxf(b0, EPS_W);
    float w1 = 1.0f / fmaxf(b1, EPS_W);
    float w2 = 1.0f / fmaxf(b2, EPS_W);
    float inv = 1.0f / (w0 + w1 + w2);
    wout[i * 3 + 0] = w0 * inv;
    wout[i * 3 + 1] = w1 * inv;
    wout[i * 3 + 2] = w2 * inv;
    iout[i * 3 + 0] = i0;
    iout[i * 3 + 1] = i1;
    iout[i * 3 + 2] = i2;
}

// ---------------------------------------------------------------------------
// Final fusion: out = relu(bn(h_main)) + sum_k w[k] * f_sub[idx[k]].
// ---------------------------------------------------------------------------
__global__ void final_kernel(const float* __restrict__ H,
                             const float* __restrict__ scale,
                             const float* __restrict__ shift,
                             const float* __restrict__ fsub,
                             const float* __restrict__ wn,
                             const int* __restrict__ kidx,
                             float* __restrict__ out)
{
    int t = blockIdx.x * blockDim.x + threadIdx.x;  // over 16384*128
    int row = t >> 7;
    int cidx = t & (NCOL - 1);
    float v = fmaf(H[t], scale[cidx], shift[cidx]);
    v = v > 0.0f ? v : 0.0f;
    v = fmaf(wn[row * 3 + 0], fsub[(size_t)kidx[row * 3 + 0] * NCOL + cidx], v);
    v = fmaf(wn[row * 3 + 1], fsub[(size_t)kidx[row * 3 + 1] * NCOL + cidx], v);
    v = fmaf(wn[row * 3 + 2], fsub[(size_t)kidx[row * 3 + 2] * NCOL + cidx], v);
    out[t] = v;
}

__global__ void zero_stats_kernel(float* __restrict__ p)
{
    p[blockIdx.x * 256 + threadIdx.x] = 0.0f;
}

// ---------------------------------------------------------------------------
// Host-side launcher (graph-capture safe: only kernel launches on `stream`).
// ---------------------------------------------------------------------------
extern "C" void kernel_launch(void* const* d_in, const int* in_sizes, int n_in,
                              void* d_out, int out_size, void* d_ws, size_t ws_size,
                              hipStream_t stream)
{
    (void)in_sizes; (void)n_in; (void)out_size; (void)ws_size;

    const float* x       = (const float*)d_in[0];   // [16384, 128]
    const float* x_sub   = (const float*)d_in[1];   // [4096, 256]
    const float* pos     = (const float*)d_in[2];   // [16384, 3]
    const float* pos_sub = (const float*)d_in[3];   // [4096, 3]
    const float* W_sub   = (const float*)d_in[4];   // [256, 128]
    const float* b_sub   = (const float*)d_in[5];
    const float* g_sub   = (const float*)d_in[6];
    const float* be_sub  = (const float*)d_in[7];
    const float* Wm      = (const float*)d_in[8];   // [128, 128]
    const float* bm      = (const float*)d_in[9];
    const float* gm      = (const float*)d_in[10];
    const float* bem     = (const float*)d_in[11];
    float* out = (float*)d_out;                      // [16384, 128]

    // Workspace layout (floats), ~11 MB total.
    float* ws        = (float*)d_ws;
    float* h_sub     = ws;                       // 4096*128   (becomes f_sub)
    float* h_main    = h_sub + 4096 * 128;       // 16384*128
    float* stats     = h_main + 16384 * 128;     // 1024 floats:
    float* sum_sub   = stats + 0 * 128;
    float* sq_sub    = stats + 1 * 128;
    float* sum_m     = stats + 2 * 128;
    float* sq_m      = stats + 3 * 128;
    float* scale_sub = stats + 4 * 128;
    float* shift_sub = stats + 5 * 128;
    float* scale_m   = stats + 6 * 128;
    float* shift_m   = stats + 7 * 128;
    float* knn_w     = stats + 8 * 128;          // 16384*3
    int*   knn_i     = (int*)(knn_w + 16384 * 3);          // 16384*3
    _Float16* wt_sub  = (_Float16*)(knn_i + 16384 * 3);    // 128*256 f16 (16B-aligned)
    _Float16* wt_main = wt_sub + 128 * 256;                // 128*128 f16

    // 1) zero BN stat accumulators (harness does not re-poison between replays)
    zero_stats_kernel<<<4, 256, 0, stream>>>(stats);

    // 2) one-time weight conversion to transposed f16
    convert_w_kernel<<<(128 * 256) / 256, 256, 0, stream>>>(W_sub, wt_sub, 256);
    convert_w_kernel<<<(128 * 128) / 256, 256, 0, stream>>>(Wm, wt_main, 128);

    // 3) f_sub branch: GEMM(4096x256 @ 256x128) + BN stats, finalize, ReLU
    gemm_bn_kernel<256><<<4096 / 64, 256, 0, stream>>>(x_sub, wt_sub, b_sub, h_sub,
                                                       sum_sub, sq_sub, 4096);
    bn_finalize_kernel<<<1, 128, 0, stream>>>(sum_sub, sq_sub, g_sub, be_sub,
                                              scale_sub, shift_sub, 1.0f / 4096.0f);
    bn_relu_kernel<<<(4096 * 128) / 256, 256, 0, stream>>>(h_sub, scale_sub,
                                                           shift_sub, 4096 * 128);

    // 4) main branch GEMM(16384x128 @ 128x128) + BN stats, finalize
    gemm_bn_kernel<128><<<16384 / 64, 256, 0, stream>>>(x, wt_main, bm, h_main,
                                                        sum_m, sq_m, 16384);
    bn_finalize_kernel<<<1, 128, 0, stream>>>(sum_m, sq_m, gm, bem,
                                              scale_m, shift_m, 1.0f / 16384.0f);

    // 5) KNN (k=3), coarse cloud LDS-resident
    knn_kernel<<<16384 / 256, 256, 0, stream>>>(pos, pos_sub, knn_w, knn_i);

    // 6) fused BN+ReLU+interpolate-add
    final_kernel<<<(16384 * 128) / 256, 256, 0, stream>>>(h_main, scale_m, shift_m,
                                                          h_sub, knn_w, knn_i, out);
}
